// NCF_27702539059306
// MI455X (gfx1250) — compile-verified, matrix operand order checked by reading, not run on hardware
//
#include <hip/hip_runtime.h>

typedef __attribute__((ext_vector_type(16))) _Float16 v16h;
typedef __attribute__((ext_vector_type(8)))  float    v8f;

#define NB    8            // batch samples
#define NPS   (1 << 22)    // elements per sample (T*C*H*W)
#define EPB   16384        // elements per streaming block
#define BPS   256          // streaming blocks per sample
#define NBLK  (NB * BPS)   // 2048 streaming blocks
#define PIX   16384        // H*W
#define EMBD  64
#define HIDD  32
#define SENT  0xFFFFFFFFu

// ---------------------------------------------------------------------------
// Kernel 1: pred[128*128] via WMMA f16 (16x16x32), GEMM1 64->32, silu, GEMM2 ->1
// 1 wave handles a 16-row tile of the 16384-row "inter" matrix.
// ---------------------------------------------------------------------------
__global__ __launch_bounds__(256) void pred_kernel(
    const float* __restrict__ xe, const float* __restrict__ ye,
    const float* __restrict__ w1, const float* __restrict__ b1,
    const float* __restrict__ w2, const float* __restrict__ b2,
    float* __restrict__ pred)
{
  __shared__ float hid[8][16][32];
  const int wave = threadIdx.x >> 5;
  const int lane = threadIdx.x & 31;
  const int tile = blockIdx.x * 8 + wave;     // 1024 tiles total
  const int p0   = tile << 4;
  const int m    = lane & 15;
  const int hi   = (lane >> 4) & 1;
  const int p    = p0 + m;
  const int hr   = p >> 7;
  const int wc   = p & 127;

  v8f acc0 = {};
  v8f acc1 = {};
  #pragma unroll
  for (int ks = 0; ks < 2; ++ks) {
    v16h a, bB0, bB1;
    #pragma unroll
    for (int j = 0; j < 8; ++j) {
      // 16-bit A-matrix 16x32 layout: lanes 0-15 K base 0, lanes 16-31 K base 8;
      // VGPR j<4 -> K = base+2j, j>=4 -> K = base+16+2(j-4); pairs (K, K+1).
      const int Klo = hi * 8 + (j < 4 ? 2 * j : 16 + 2 * (j - 4));
      const int e   = ks * 32 + Klo;
      a[2 * j]     = (_Float16)(xe[hr * EMBD + e]     * ye[wc * EMBD + e]);
      a[2 * j + 1] = (_Float16)(xe[hr * EMBD + e + 1] * ye[wc * EMBD + e + 1]);
      const int c = lane & 15;                // B: lane holds column n
      bB0[2 * j]     = (_Float16)w1[e * HIDD + c];
      bB0[2 * j + 1] = (_Float16)w1[(e + 1) * HIDD + c];
      bB1[2 * j]     = (_Float16)w1[e * HIDD + 16 + c];
      bB1[2 * j + 1] = (_Float16)w1[(e + 1) * HIDD + 16 + c];
    }
    acc0 = __builtin_amdgcn_wmma_f32_16x16x32_f16(false, a, false, bB0,
                                                  (short)0, acc0, false, false);
    acc1 = __builtin_amdgcn_wmma_f32_16x16x32_f16(false, a, false, bB1,
                                                  (short)0, acc1, false, false);
  }
  // bias + silu, scatter C layout (VGPR r: lanes0-15 M=r, lanes16-31 M=r+8) to LDS
  #pragma unroll
  for (int r = 0; r < 8; ++r) {
    const int row = hi ? r + 8 : r;
    const int c   = lane & 15;
    const float v0 = acc0[r] + b1[c];
    const float v1 = acc1[r] + b1[16 + c];
    hid[wave][row][c]      = v0 / (1.0f + __expf(-v0));
    hid[wave][row][16 + c] = v1 / (1.0f + __expf(-v1));
  }
  __syncthreads();
  if (lane < 16) {                            // GEMM2: N=1, plain dot
    float s = b2[0];
    #pragma unroll
    for (int c = 0; c < 32; ++c) s += hid[wave][lane][c] * w2[c];
    pred[p0 + lane] = s;
  }
}

// ---------------------------------------------------------------------------
// Kernel 2: level-1 histogram (bits[31:20], 4096 bins) + per-sample mask sum
// ---------------------------------------------------------------------------
__global__ __launch_bounds__(256) void hist1_kernel(
    const float* __restrict__ rnd, const float* __restrict__ mask,
    unsigned* __restrict__ hist, float* __restrict__ maskSum)
{
  __shared__ unsigned h[4096];
  __shared__ float red[256];
  const int tid = threadIdx.x;
  for (int i = tid; i < 4096; i += 256) h[i] = 0u;
  __syncthreads();
  const int b     = blockIdx.x >> 8;
  const int chunk = blockIdx.x & 255;
  const long long base = (long long)b * NPS + (long long)chunk * EPB;
  const float4* r4 = (const float4*)(rnd + base);
  const float4* m4 = (const float4*)(mask + base);
  float ms = 0.f;
  for (int it = 0; it < 16; ++it) {
    const int j = it * 256 + tid;
    const float4 rv = r4[j];
    const float4 mv = m4[j];
    ms += mv.x + mv.y + mv.z + mv.w;
    atomicAdd(&h[__float_as_uint(rv.x * mv.x) >> 20], 1u);
    atomicAdd(&h[__float_as_uint(rv.y * mv.y) >> 20], 1u);
    atomicAdd(&h[__float_as_uint(rv.z * mv.z) >> 20], 1u);
    atomicAdd(&h[__float_as_uint(rv.w * mv.w) >> 20], 1u);
  }
  __syncthreads();
  for (int i = tid; i < 4096; i += 256) {
    const unsigned c = h[i];
    if (c) atomicAdd(&hist[b * 4096 + i], c);
  }
  red[tid] = ms;
  __syncthreads();
  for (int s = 128; s > 0; s >>= 1) {
    if (tid < s) red[tid] += red[tid + s];
    __syncthreads();
  }
  if (tid == 0) atomicAdd(&maskSum[b], red[0]);
}

// ---------------------------------------------------------------------------
// Kernel 3: k = round(0.1 * num_observed) per sample (banker's rounding = rintf)
// ---------------------------------------------------------------------------
__global__ void kcalc_kernel(const float* __restrict__ maskSum,
                             unsigned* __restrict__ kArr)
{
  const int b = threadIdx.x;
  if (b >= NB) return;
  kArr[b] = (unsigned)(long long)rintf(maskSum[b] * 0.1f);
}

// ---------------------------------------------------------------------------
// Kernel 4: generic radix-select step: walk histogram descending, find the bin
// containing the k-th largest; emit refined prefix + remaining need.
// ---------------------------------------------------------------------------
__global__ void select_kernel(const unsigned* __restrict__ hist, int bins,
                              int outShift, const unsigned* __restrict__ prefixIn,
                              const unsigned* __restrict__ needIn,
                              unsigned* __restrict__ prefixOut,
                              unsigned* __restrict__ needOut)
{
  const int b = threadIdx.x;
  if (b >= NB) return;
  const unsigned k   = needIn[b];
  const unsigned pin = prefixIn ? prefixIn[b] : 0u;
  if (k == 0u || pin == SENT) { prefixOut[b] = SENT; needOut[b] = 0u; return; }
  unsigned acc = 0u, sel = 0u, nd = 0u;
  for (int v = bins - 1; v >= 0; --v) {
    const unsigned c = hist[b * bins + v];
    if (acc + c >= k) { sel = (unsigned)v; nd = k - acc; break; }
    acc += c;
  }
  prefixOut[b] = (pin << outShift) | sel;
  needOut[b]   = nd;
}

// ---------------------------------------------------------------------------
// Kernel 5: refinement histogram among elements matching current prefix
// ---------------------------------------------------------------------------
__global__ __launch_bounds__(256) void hist_refine_kernel(
    const float* __restrict__ rnd, const float* __restrict__ mask,
    const unsigned* __restrict__ prefix, int matchShift, int binShift, int bins,
    unsigned* __restrict__ histOut)
{
  __shared__ unsigned h[4096];
  const int tid = threadIdx.x;
  for (int i = tid; i < bins; i += 256) h[i] = 0u;
  __syncthreads();
  const int b = blockIdx.x >> 8;
  const unsigned pref = prefix[b];
  if (pref != SENT) {
    const int chunk = blockIdx.x & 255;
    const long long base = (long long)b * NPS + (long long)chunk * EPB;
    const float4* r4 = (const float4*)(rnd + base);
    const float4* m4 = (const float4*)(mask + base);
    const unsigned bm = (unsigned)(bins - 1);
    for (int it = 0; it < 16; ++it) {
      const int j = it * 256 + tid;
      const float4 rv = r4[j];
      const float4 mv = m4[j];
      unsigned bt;
      bt = __float_as_uint(rv.x * mv.x);
      if ((bt >> matchShift) == pref) atomicAdd(&h[(bt >> binShift) & bm], 1u);
      bt = __float_as_uint(rv.y * mv.y);
      if ((bt >> matchShift) == pref) atomicAdd(&h[(bt >> binShift) & bm], 1u);
      bt = __float_as_uint(rv.z * mv.z);
      if ((bt >> matchShift) == pref) atomicAdd(&h[(bt >> binShift) & bm], 1u);
      bt = __float_as_uint(rv.w * mv.w);
      if ((bt >> matchShift) == pref) atomicAdd(&h[(bt >> binShift) & bm], 1u);
    }
  }
  __syncthreads();
  for (int i = tid; i < bins; i += 256) {
    const unsigned c = h[i];
    if (c) atomicAdd(&histOut[b * bins + i], c);
  }
}

// ---------------------------------------------------------------------------
// Kernel 6: collect indices of elements whose rm-bits equal the threshold T
// ---------------------------------------------------------------------------
__global__ __launch_bounds__(256) void eq_gather_kernel(
    const float* __restrict__ rnd, const float* __restrict__ mask,
    const unsigned* __restrict__ Tarr, const unsigned* __restrict__ needEq,
    unsigned* __restrict__ eqCount, unsigned* __restrict__ eqBuf)
{
  const int b = blockIdx.x >> 8;
  if (needEq[b] == 0u) return;          // uniform per block
  const unsigned t = Tarr[b];
  const int chunk = blockIdx.x & 255;
  const long long base = (long long)b * NPS + (long long)chunk * EPB;
  const float4* r4 = (const float4*)(rnd + base);
  const float4* m4 = (const float4*)(mask + base);
  for (int it = 0; it < 16; ++it) {
    const int j = it * 256 + threadIdx.x;
    const float4 rv = r4[j];
    const float4 mv = m4[j];
    const unsigned li = (unsigned)(chunk * EPB + j * 4);
    const float rr[4] = {rv.x, rv.y, rv.z, rv.w};
    const float mm[4] = {mv.x, mv.y, mv.z, mv.w};
    #pragma unroll
    for (int c = 0; c < 4; ++c) {
      if (__float_as_uint(rr[c] * mm[c]) == t) {
        const unsigned slot = atomicAdd(&eqCount[b], 1u);
        if (slot < 4096u) eqBuf[b * 4096 + slot] = li + (unsigned)c;
      }
    }
  }
}

// ---------------------------------------------------------------------------
// Kernel 7: stable-tie cutoff = index of rank-E element among equals
// (elements equal to T with idx < cutoff are in the top-k, matching stable argsort)
// ---------------------------------------------------------------------------
__global__ __launch_bounds__(256) void cutoff_kernel(
    const unsigned* __restrict__ needEq, const unsigned* __restrict__ eqCount,
    const unsigned* __restrict__ eqBuf, unsigned* __restrict__ cutoff)
{
  __shared__ unsigned sbuf[4096];
  __shared__ unsigned result;
  const int b = blockIdx.x;
  const unsigned E = needEq[b];
  const unsigned n = min(eqCount[b], 4096u);
  if (threadIdx.x == 0) result = 0u;
  for (int i = threadIdx.x; i < (int)n; i += 256) sbuf[i] = eqBuf[b * 4096 + i];
  __syncthreads();
  if (E > 0u) {
    if (E >= n) {
      if (threadIdx.x == 0) result = SENT;   // all equals are in top-k
    } else {
      for (int i = threadIdx.x; i < (int)n; i += 256) {
        const unsigned vi = sbuf[i];
        unsigned rank = 0u;
        for (int j = 0; j < (int)n; ++j) rank += (sbuf[j] < vi) ? 1u : 0u;
        if (rank == E) result = vi;          // unique writer
      }
    }
  }
  __syncthreads();
  if (threadIdx.x == 0) cutoff[b] = result;
}

// ---------------------------------------------------------------------------
// Kernel 8: fused loss pass: target = mask - keep, accumulate sum((d-pred)*t)^2
// and sum(t), with deterministic per-block tree reduction.
// ---------------------------------------------------------------------------
__global__ __launch_bounds__(256) void loss_kernel(
    const float* __restrict__ data, const float* __restrict__ mask,
    const float* __restrict__ rnd, const float* __restrict__ pred,
    const unsigned* __restrict__ Tarr, const unsigned* __restrict__ cutArr,
    float* __restrict__ pSq, float* __restrict__ pCnt)
{
  __shared__ float rs[256], rc[256];
  const int tid   = threadIdx.x;
  const int b     = blockIdx.x >> 8;
  const int chunk = blockIdx.x & 255;
  const unsigned t   = Tarr[b];
  const unsigned cut = cutArr[b];
  const long long base = (long long)b * NPS + (long long)chunk * EPB;
  const float4* d4 = (const float4*)(data + base);
  const float4* m4 = (const float4*)(mask + base);
  const float4* r4 = (const float4*)(rnd + base);
  float sq = 0.f, cn = 0.f;
  for (int it = 0; it < 16; ++it) {
    const int j = it * 256 + tid;
    const float4 dv = d4[j];
    const float4 mv = m4[j];
    const float4 rv = r4[j];
    const unsigned li = (unsigned)(chunk * EPB + j * 4);
    const float dd[4] = {dv.x, dv.y, dv.z, dv.w};
    const float mm[4] = {mv.x, mv.y, mv.z, mv.w};
    const float rr[4] = {rv.x, rv.y, rv.z, rv.w};
    #pragma unroll
    for (int c = 0; c < 4; ++c) {
      const float rm = rr[c] * mm[c];
      const unsigned bits = __float_as_uint(rm);
      const unsigned idx  = li + (unsigned)c;
      const bool topk = (bits > t) || (bits == t && idx < cut);
      const float keep = (rm > 0.f && !topk) ? 1.f : 0.f;
      const float targ = mm[c] - keep;
      const float res  = (dd[c] - pred[idx & (PIX - 1)]) * targ;
      sq += res * res;
      cn += targ;
    }
  }
  rs[tid] = sq;
  rc[tid] = cn;
  __syncthreads();
  for (int s = 128; s > 0; s >>= 1) {
    if (tid < s) { rs[tid] += rs[tid + s]; rc[tid] += rc[tid + s]; }
    __syncthreads();
  }
  if (tid == 0) { pSq[blockIdx.x] = rs[0]; pCnt[blockIdx.x] = rc[0]; }
}

// ---------------------------------------------------------------------------
// Kernel 9: deterministic final reduction + divide
// ---------------------------------------------------------------------------
__global__ __launch_bounds__(256) void finalize_kernel(
    const float* __restrict__ pSq, const float* __restrict__ pCnt, int n,
    float* __restrict__ out)
{
  __shared__ float rs[256], rc[256];
  const int tid = threadIdx.x;
  float a = 0.f, c = 0.f;
  for (int i = tid; i < n; i += 256) { a += pSq[i]; c += pCnt[i]; }
  rs[tid] = a;
  rc[tid] = c;
  __syncthreads();
  for (int s = 128; s > 0; s >>= 1) {
    if (tid < s) { rs[tid] += rs[tid + s]; rc[tid] += rc[tid + s]; }
    __syncthreads();
  }
  if (tid == 0) {
    const float num = rc[0];
    out[0] = rs[0] / (num > 0.f ? num : 1.0f);
  }
}

// ---------------------------------------------------------------------------
extern "C" void kernel_launch(void* const* d_in, const int* in_sizes, int n_in,
                              void* d_out, int out_size, void* d_ws, size_t ws_size,
                              hipStream_t stream)
{
  (void)in_sizes; (void)n_in; (void)out_size; (void)ws_size;
  const float* data = (const float*)d_in[0];
  const float* mask = (const float*)d_in[1];
  const float* rnd  = (const float*)d_in[4];
  const float* xe   = (const float*)d_in[5];
  const float* ye   = (const float*)d_in[6];
  const float* w1   = (const float*)d_in[7];
  const float* b1   = (const float*)d_in[8];
  const float* w2   = (const float*)d_in[9];
  const float* b2   = (const float*)d_in[10];

  unsigned char* w = (unsigned char*)d_ws;
  size_t off = 0;
  float* pred = (float*)(w + off);          off += PIX * 4;
  const size_t zeroStart = off;
  unsigned* hist1   = (unsigned*)(w + off); off += NB * 4096 * 4;
  unsigned* hist2   = (unsigned*)(w + off); off += NB * 4096 * 4;
  unsigned* hist3   = (unsigned*)(w + off); off += NB * 256 * 4;
  float*    maskSum = (float*)(w + off);    off += 256;
  unsigned* eqCount = (unsigned*)(w + off); off += 256;
  const size_t zeroBytes = off - zeroStart;
  unsigned* kArr   = (unsigned*)(w + off);  off += 256;
  unsigned* pref1  = (unsigned*)(w + off);  off += 256;
  unsigned* need1  = (unsigned*)(w + off);  off += 256;
  unsigned* pref2  = (unsigned*)(w + off);  off += 256;
  unsigned* need2  = (unsigned*)(w + off);  off += 256;
  unsigned* Tarr   = (unsigned*)(w + off);  off += 256;
  unsigned* needEq = (unsigned*)(w + off);  off += 256;
  unsigned* cutA   = (unsigned*)(w + off);  off += 256;
  unsigned* eqBuf  = (unsigned*)(w + off);  off += NB * 4096 * 4;
  float*    pSq    = (float*)(w + off);     off += NBLK * 4;
  float*    pCnt   = (float*)(w + off);     off += NBLK * 4;

  hipMemsetAsync(w + zeroStart, 0, zeroBytes, stream);

  pred_kernel<<<PIX / 16 / 8, 256, 0, stream>>>(xe, ye, w1, b1, w2, b2, pred);
  hist1_kernel<<<NBLK, 256, 0, stream>>>(rnd, mask, hist1, maskSum);
  kcalc_kernel<<<1, 32, 0, stream>>>(maskSum, kArr);
  select_kernel<<<1, 32, 0, stream>>>(hist1, 4096, 12, nullptr, kArr, pref1, need1);
  hist_refine_kernel<<<NBLK, 256, 0, stream>>>(rnd, mask, pref1, 20, 8, 4096, hist2);
  select_kernel<<<1, 32, 0, stream>>>(hist2, 4096, 12, pref1, need1, pref2, need2);
  hist_refine_kernel<<<NBLK, 256, 0, stream>>>(rnd, mask, pref2, 8, 0, 256, hist3);
  select_kernel<<<1, 32, 0, stream>>>(hist3, 256, 8, pref2, need2, Tarr, needEq);
  eq_gather_kernel<<<NBLK, 256, 0, stream>>>(rnd, mask, Tarr, needEq, eqCount, eqBuf);
  cutoff_kernel<<<NB, 256, 0, stream>>>(needEq, eqCount, eqBuf, cutA);
  loss_kernel<<<NBLK, 256, 0, stream>>>(data, mask, rnd, pred, Tarr, cutA, pSq, pCnt);
  finalize_kernel<<<1, 256, 0, stream>>>(pSq, pCnt, NBLK, (float*)d_out);
}
